// Model_11527692222992
// MI455X (gfx1250) — matrix operand order repaired and checked by reading, not compiled, not run on hardware
//
#include <hip/hip_runtime.h>

#define B_ 8192
#define D_ 3072
#define C_ 512
#define E_ 8

typedef float v8f  __attribute__((ext_vector_type(8)));
typedef __bf16 v16bf __attribute__((ext_vector_type(16)));
typedef __bf16 v8bf  __attribute__((ext_vector_type(8)));
typedef __bf16 v4bf  __attribute__((ext_vector_type(4)));

union bfpack { v16bf v16; v8bf v8[2]; };

// ---------------- routing: counting sort of rows by label ----------------

__global__ void k_zero(int* cw) {
    if (threadIdx.x < 64) cw[threadIdx.x] = 0;
}

__global__ void k_hist(const int* __restrict__ label, int* __restrict__ counts) {
    int b = blockIdx.x * blockDim.x + threadIdx.x;
    if (b < B_) atomicAdd(&counts[label[b] & (E_ - 1)], 1);
}

__global__ void k_scan(const int* __restrict__ counts, int* __restrict__ offsets,
                       int* __restrict__ cursor) {
    int acc = 0;
    for (int e = 0; e < E_; ++e) { offsets[e] = acc; cursor[e] = acc; acc += counts[e]; }
    offsets[E_] = acc;
}

__global__ void k_scatter(const int* __restrict__ label, int* __restrict__ cursor,
                          int* __restrict__ row_idx) {
    int b = blockIdx.x * blockDim.x + threadIdx.x;
    if (b < B_) {
        int p = atomicAdd(&cursor[label[b] & (E_ - 1)], 1);
        row_idx[p] = b;
    }
}

// ---------------- helpers ----------------

// x = hi + lo with hi = bf16(x), lo = bf16(x - hi): ~16 effective mantissa bits.
__device__ __forceinline__ void split4(const float4 v, v4bf& h, v4bf& l) {
    float xs[4] = {v.x, v.y, v.z, v.w};
    v4bf hh, ll;
#pragma unroll
    for (int j = 0; j < 4; ++j) {
        __bf16 hb = (__bf16)xs[j];
        hh[j] = hb;
        ll[j] = (__bf16)(xs[j] - (float)hb);
    }
    h = hh; l = ll;
}

__device__ __forceinline__ v16bf ld_frag(const __bf16* p0, const __bf16* p1) {
    bfpack u;
    u.v8[0] = *(const v8bf*)p0;
    u.v8[1] = *(const v8bf*)p1;
    return u.v16;
}

// ---------- grouped GEMM: f32 via bf16x3 on V_WMMA_F32_16X16X32_BF16 ----------
// Out[m, n] = act( sum_k A[m, k] * W[e, n, k] + bias[e, n] )
// Block = 128 threads (4 waves), 64x64 output tile, K-tile = 32, LDS double-buffered.

template <int N_TOT, int K_TOT, bool RELU, bool GATHER_A, bool SCATTER_OUT>
__global__ __launch_bounds__(128)
void k_gemm(const float* __restrict__ A,      // [B, K_TOT] (img or code)
            const float* __restrict__ W,      // [E, N_TOT, K_TOT]
            const float* __restrict__ bias,   // [E, N_TOT]
            const int* __restrict__ counts,
            const int* __restrict__ offsets,
            const int* __restrict__ row_idx,
            float* __restrict__ Out)          // [B, N_TOT]
{
    const int e   = blockIdx.z;
    const int cnt = counts[e];
    const int m0  = blockIdx.y * 64;
    if (m0 >= cnt) return;                    // tile past this expert's rows
    const int base = offsets[e];
    const int n0   = blockIdx.x * 64;

    constexpr int SR = 40;                    // LDS row stride (bf16): 80 B, 16B-aligned rows
    __shared__ __bf16 Ahi[2][64][SR];
    __shared__ __bf16 Alo[2][64][SR];
    __shared__ __bf16 Bhi[2][64][SR];
    __shared__ __bf16 Blo[2][64][SR];

    const int tid  = threadIdx.x;
    const int lane = tid & 31;
    const int wv   = tid >> 5;                // wave -> M stripe 16*wv
    const int half = lane >> 4;
    const int l15  = lane & 15;
    const int h8   = half * 8;                // A frag K-chunk select
    const int h16  = half * 16;               // unused: B frags also select K-chunks with h8

    v8f acc[4];
#pragma unroll
    for (int ns = 0; ns < 4; ++ns)
#pragma unroll
        for (int i = 0; i < 8; ++i) acc[ns][i] = 0.0f;

    // Cooperative-load pointers: idx = it*128 + tid covers 64 rows x 8 float4 per tile.
    const float* aptr[4];
    const float* bptr[4];
    const float* Wb = W + ((size_t)e * N_TOT + n0) * (size_t)K_TOT;
#pragma unroll
    for (int it = 0; it < 4; ++it) {
        int idx = it * 128 + tid;
        int i   = idx >> 3;
        int ch  = (idx & 7) * 4;
        int mi  = m0 + i;
        int meff = (mi < cnt) ? mi : (cnt - 1);   // clamp ragged edge (masked at store)
        size_t srow = GATHER_A ? (size_t)row_idx[base + meff] : (size_t)(base + meff);
        aptr[it] = A  + srow * (size_t)K_TOT + ch;
        bptr[it] = Wb + (size_t)i * (size_t)K_TOT + ch;
    }

    // prologue: load + split + store tile 0 into buffer 0
    float4 avr[4], bvr[4];
#pragma unroll
    for (int it = 0; it < 4; ++it) {
        avr[it] = *(const float4*)(aptr[it]);
        bvr[it] = *(const float4*)(bptr[it]);
    }
#pragma unroll
    for (int it = 0; it < 4; ++it) {
        int idx = it * 128 + tid;
        int i   = idx >> 3;
        int ch  = (idx & 7) * 4;
        v4bf h, l;
        split4(avr[it], h, l);
        *(v4bf*)&Ahi[0][i][ch] = h;
        *(v4bf*)&Alo[0][i][ch] = l;
        split4(bvr[it], h, l);
        *(v4bf*)&Bhi[0][i][ch] = h;
        *(v4bf*)&Blo[0][i][ch] = l;
    }
    __syncthreads();

    constexpr int NT = K_TOT / 32;
    for (int t = 0; t < NT; ++t) {
        const int buf = t & 1;

        // issue next K-tile's global loads before compute (overlap with WMMA)
        if (t + 1 < NT) {
            const int kn = (t + 1) * 32;
#pragma unroll
            for (int it = 0; it < 4; ++it) {
                avr[it] = *(const float4*)(aptr[it] + kn);
                bvr[it] = *(const float4*)(bptr[it] + kn);
            }
        }

        // compute: 12 WMMAs per wave per K-tile (hi*hi + hi*lo + lo*hi)
        {
            const int m = 16 * wv + l15;
            v16bf ah = ld_frag(&Ahi[buf][m][h8], &Ahi[buf][m][16 + h8]);
            v16bf al = ld_frag(&Alo[buf][m][h8], &Alo[buf][m][16 + h8]);
#pragma unroll
            for (int ns = 0; ns < 4; ++ns) {
                const int n = ns * 16 + l15;
                v16bf bh = ld_frag(&Bhi[buf][n][h8], &Bhi[buf][n][16 + h8]);
                v16bf bl = ld_frag(&Blo[buf][n][h8], &Blo[buf][n][16 + h8]);
                acc[ns] = __builtin_amdgcn_wmma_f32_16x16x32_bf16(
                    false, ah, false, bl, (short)0, acc[ns], false, false);
                acc[ns] = __builtin_amdgcn_wmma_f32_16x16x32_bf16(
                    false, al, false, bh, (short)0, acc[ns], false, false);
                acc[ns] = __builtin_amdgcn_wmma_f32_16x16x32_bf16(
                    false, ah, false, bh, (short)0, acc[ns], false, false);
            }
        }

        if (t + 1 < NT) {
            __syncthreads();                  // all waves done reading buf^1 (iter t-1)
#pragma unroll
            for (int it = 0; it < 4; ++it) {
                int idx = it * 128 + tid;
                int i   = idx >> 3;
                int ch  = (idx & 7) * 4;
                v4bf h, l;
                split4(avr[it], h, l);
                *(v4bf*)&Ahi[buf ^ 1][i][ch] = h;
                *(v4bf*)&Alo[buf ^ 1][i][ch] = l;
                split4(bvr[it], h, l);
                *(v4bf*)&Bhi[buf ^ 1][i][ch] = h;
                *(v4bf*)&Blo[buf ^ 1][i][ch] = l;
            }
            __syncthreads();
        }
    }

    // Epilogue: C/D layout -> VGPR i holds M=i (lanes 0-15) and M=i+8 (lanes 16-31)
    float bv[4];
#pragma unroll
    for (int ns = 0; ns < 4; ++ns)
        bv[ns] = bias[(size_t)e * N_TOT + n0 + ns * 16 + l15];

#pragma unroll
    for (int i = 0; i < 8; ++i) {
        int mt = 16 * wv + i + 8 * half;
        int m  = m0 + mt;
        if (m < cnt) {
            size_t orow = SCATTER_OUT ? (size_t)row_idx[base + m] : (size_t)(base + m);
            float* orow_p = Out + orow * (size_t)N_TOT + n0;
#pragma unroll
            for (int ns = 0; ns < 4; ++ns) {
                float v = acc[ns][i] + bv[ns];
                if (RELU) v = fmaxf(v, 0.0f);
                orow_p[ns * 16 + l15] = v;
            }
        }
    }
}

// ---------------- deterministic MSE loss ----------------

__global__ void k_loss1(const float* __restrict__ img, const float* __restrict__ dec,
                        float* __restrict__ partial) {
    __shared__ float red[256];
    const size_t n = (size_t)B_ * D_;
    float s = 0.0f;
    for (size_t i = (size_t)blockIdx.x * blockDim.x + threadIdx.x; i < n;
         i += (size_t)gridDim.x * blockDim.x) {
        float d = dec[i] - img[i];
        s += d * d;
    }
    red[threadIdx.x] = s;
    __syncthreads();
    for (int w = 128; w > 0; w >>= 1) {
        if ((int)threadIdx.x < w) red[threadIdx.x] += red[threadIdx.x + w];
        __syncthreads();
    }
    if (threadIdx.x == 0) partial[blockIdx.x] = red[0];
}

__global__ void k_loss2(const float* __restrict__ partial, float* __restrict__ out) {
    __shared__ float red[1024];
    red[threadIdx.x] = partial[threadIdx.x];
    __syncthreads();
    for (int w = 512; w > 0; w >>= 1) {
        if ((int)threadIdx.x < w) red[threadIdx.x] += red[threadIdx.x + w];
        __syncthreads();
    }
    if (threadIdx.x == 0) out[0] = red[0] / (float)((size_t)B_ * D_);
}

// ---------------- launch ----------------

extern "C" void kernel_launch(void* const* d_in, const int* in_sizes, int n_in,
                              void* d_out, int out_size, void* d_ws, size_t ws_size,
                              hipStream_t stream) {
    const float* img   = (const float*)d_in[0];   // [B, D]
    const int*   label = (const int*)d_in[1];     // [B]
    const float* W_enc = (const float*)d_in[2];   // [E, C, D]
    const float* b_enc = (const float*)d_in[3];   // [E, C]
    const float* W_dec = (const float*)d_in[4];   // [E, D, C]
    const float* b_dec = (const float*)d_in[5];   // [E, D]

    float* out     = (float*)d_out;
    float* loss    = out;                          // d_out[0]
    float* decoded = out + 1;                      // d_out[1..] = [B, D]

    // workspace layout
    int* cw      = (int*)d_ws;
    int* counts  = cw;                             // [0..7]
    int* offsets = cw + 8;                         // [8..16]
    int* cursor  = cw + 24;                        // [24..31]
    int* row_idx = cw + 64;                        // [64..64+B)
    float* code    = (float*)((char*)d_ws + 65536);       // [B, C] = 16 MB
    float* partial = code + (size_t)B_ * C_;              // [1024]

    k_zero<<<1, 64, 0, stream>>>(cw);
    k_hist<<<B_ / 256, 256, 0, stream>>>(label, counts);
    k_scan<<<1, 1, 0, stream>>>(counts, offsets, cursor);
    k_scatter<<<B_ / 256, 256, 0, stream>>>(label, cursor, row_idx);

    // Encoder: code[p, c] = relu(img[row_idx[p]] . W_enc[e, c, :] + b_enc[e, c])
    k_gemm<C_, D_, /*RELU=*/true, /*GATHER_A=*/true, /*SCATTER_OUT=*/false>
        <<<dim3(C_ / 64, B_ / 64, E_), 128, 0, stream>>>(
            img, W_enc, b_enc, counts, offsets, row_idx, code);

    // Decoder: decoded[row_idx[p], d] = code[p] . W_dec[e, d, :] + b_dec[e, d]
    k_gemm<D_, C_, /*RELU=*/false, /*GATHER_A=*/false, /*SCATTER_OUT=*/true>
        <<<dim3(D_ / 64, B_ / 64, E_), 128, 0, stream>>>(
            code, W_dec, b_dec, counts, offsets, row_idx, decoded);

    k_loss1<<<1024, 256, 0, stream>>>(img, decoded, partial);
    k_loss2<<<1, 1024, 0, stream>>>(partial, loss);
}
